// GIN_76690936037967
// MI455X (gfx1250) — matrix operand order repaired and checked by reading, not compiled, not run on hardware
//
#include <hip/hip_runtime.h>
#include <hip/hip_bf16.h>

typedef __bf16 bf16_t;
typedef __attribute__((ext_vector_type(8)))  __bf16 bf16x8;
typedef __attribute__((ext_vector_type(16))) __bf16 bf16x16;
typedef __attribute__((ext_vector_type(8)))  float  f32x8;

// ---------------------------------------------------------------------------
// Utility kernels
// ---------------------------------------------------------------------------
__global__ void zero_f32(float* p, int n) {
  int i = blockIdx.x * blockDim.x + threadIdx.x;
  if (i < n) p[i] = 0.0f;
}

__global__ void copy_f32_vec4(const float* __restrict__ a, float* __restrict__ b, int n4) {
  int i = blockIdx.x * blockDim.x + threadIdx.x;
  if (i < n4) ((float4*)b)[i] = ((const float4*)a)[i];
}

// W [K x N] fp32 -> WT [N x K] bf16 (transposed)
__global__ void cvt_f32_bf16_T(const float* __restrict__ a, bf16_t* __restrict__ b,
                               int K, int N) {
  int i = blockIdx.x * blockDim.x + threadIdx.x;
  if (i >= K * N) return;
  int k = i / N, n = i - k * N;
  b[(size_t)n * K + k] = (bf16_t)a[i];
}

// W1a [14 x 512] fp32 -> padded transposed [512 x 32] bf16 (k = 14..31 zero)
__global__ void cvt_pad_w1a_T(const float* __restrict__ w, bf16_t* __restrict__ o) {
  int i = blockIdx.x * blockDim.x + threadIdx.x;
  if (i >= 512 * 32) return;
  int n = i >> 5, k = i & 31;
  o[i] = (k < 14) ? (bf16_t)w[k * 512 + n] : (bf16_t)0.0f;
}

// h0[N x 32] = concat(x[N x 11], pos[N x 3], zeros)
__global__ void concat_pad(const float* __restrict__ x, const float* __restrict__ pos,
                           float* __restrict__ h0, int N) {
  int i = blockIdx.x * blockDim.x + threadIdx.x;
  if (i >= N * 32) return;
  int n = i >> 5, c = i & 31;
  float v = 0.0f;
  if (c < 11)      v = x[(size_t)n * 11 + c];
  else if (c < 14) v = pos[(size_t)n * 3 + (c - 11)];
  h0[i] = v;
}

// z[dst] += h[src] over all edges (z pre-initialized to h)
__global__ void edge_scatter(const float* __restrict__ h, const int* __restrict__ src,
                             const int* __restrict__ dst, float* __restrict__ z,
                             int E, int Hc) {
  int vec = Hc >> 2;                       // float4 chunks per row
  int idx = blockIdx.x * blockDim.x + threadIdx.x;
  if (idx >= E * vec) return;
  int e = idx / vec;
  int c4 = (idx - e * vec) << 2;
  int s = src[e], d = dst[e];
  float4 v = *(const float4*)&h[(size_t)s * Hc + c4];
  float* zp = &z[(size_t)d * Hc + c4];
  atomicAdd(zp + 0, v.x);
  atomicAdd(zp + 1, v.y);
  atomicAdd(zp + 2, v.z);
  atomicAdd(zp + 3, v.w);
}

// Per-column mean/var -> scale/shift (one block per column)
__global__ void bn_stats(const float* __restrict__ u, const float* __restrict__ g,
                         const float* __restrict__ be, float* __restrict__ scale,
                         float* __restrict__ shift, int M, int Nc) {
  int c = blockIdx.x;
  float s = 0.0f, s2 = 0.0f;
  for (int n = threadIdx.x; n < M; n += blockDim.x) {
    float v = u[(size_t)n * Nc + c];
    s += v; s2 += v * v;
  }
  __shared__ float r1[256];
  __shared__ float r2[256];
  r1[threadIdx.x] = s; r2[threadIdx.x] = s2;
  __syncthreads();
  for (int o = 128; o > 0; o >>= 1) {
    if ((int)threadIdx.x < o) { r1[threadIdx.x] += r1[threadIdx.x + o]; r2[threadIdx.x] += r2[threadIdx.x + o]; }
    __syncthreads();
  }
  if (threadIdx.x == 0) {
    float mean = r1[0] / (float)M;
    float var  = r2[0] / (float)M - mean * mean;
    float sc = g[c] * rsqrtf(var + 1e-5f);
    scale[c] = sc;
    shift[c] = be[c] - mean * sc;
  }
}

// h = relu(u*scale+shift); also overwrite u with h (pre-init of z for next layer)
__global__ void bn_apply_relu(float* __restrict__ u, const float* __restrict__ scale,
                              const float* __restrict__ shift, float* __restrict__ h,
                              int total) {
  int i = blockIdx.x * blockDim.x + threadIdx.x;
  if (i >= total) return;
  int c = i & 511;
  float v = fmaxf(u[i] * scale[c] + shift[c], 0.0f);
  h[i] = v;
  u[i] = v;
}

__global__ void pool_count(const int* __restrict__ batch, float* __restrict__ counts, int N) {
  int i = blockIdx.x * blockDim.x + threadIdx.x;
  if (i < N) atomicAdd(&counts[batch[i]], 1.0f);
}

__global__ void pool_sum(const float* __restrict__ h, const int* __restrict__ batch,
                         float* __restrict__ sums, int N) {
  int idx = blockIdx.x * blockDim.x + threadIdx.x;   // N * 128 (float4 chunks)
  if (idx >= N * 128) return;
  int n = idx >> 7;
  int c4 = (idx & 127) << 2;
  int g = batch[n];
  float4 v = *(const float4*)&h[(size_t)n * 512 + c4];
  float* sp = &sums[(size_t)g * 512 + c4];
  atomicAdd(sp + 0, v.x);
  atomicAdd(sp + 1, v.y);
  atomicAdd(sp + 2, v.z);
  atomicAdd(sp + 3, v.w);
}

__global__ void pool_div(const float* __restrict__ sums, const float* __restrict__ counts,
                         float* __restrict__ pooled, int total) {
  int i = blockIdx.x * blockDim.x + threadIdx.x;
  if (i >= total) return;
  pooled[i] = sums[i] / fmaxf(counts[i >> 9], 1.0f);
}

// out[g] = dot(q[g,:512], w2) + b2
__global__ void head2(const float* __restrict__ q, const float* __restrict__ w2,
                      const float* __restrict__ b2, float* __restrict__ out) {
  int g = blockIdx.x;
  int t = threadIdx.x;                       // 128 threads
  float s = 0.0f;
  for (int c = t; c < 512; c += 128) s += q[(size_t)g * 512 + c] * w2[c];
  __shared__ float red[128];
  red[t] = s;
  __syncthreads();
  for (int o = 64; o > 0; o >>= 1) {
    if (t < o) red[t] += red[t + o];
    __syncthreads();
  }
  if (t == 0) out[g] = red[0] + b2[0];
}

// ---------------------------------------------------------------------------
// bf16 WMMA GEMM:  C[M x N] = relu?( A[M x K] * W + bias )
// A fp32 (converted during LDS staging). WT is the bf16 TRANSPOSED weight,
// layout [N x K] row-major, so LDS B tile rows are contiguous in global and
// staged with GLOBAL_LOAD_ASYNC_TO_LDS_B128 (ASYNCcnt-tracked, CDNA5 path).
// N % 64 == 0, K % 32 == 0.
// Block = 256 threads = 8 waves; block tile 128x64; wave tile 32x32 (2x2 WMMA).
// ---------------------------------------------------------------------------
template <bool RELU>
__global__ __launch_bounds__(256) void gemm_bf16_wmma(
    const float* __restrict__ A, const bf16_t* __restrict__ WT,
    const float* __restrict__ bias, float* __restrict__ C,
    int M, int K, int N) {
  __shared__ __attribute__((aligned(32))) bf16_t sA[128 * 32];  // [row][k]
  __shared__ __attribute__((aligned(32))) bf16_t sB[64 * 32];   // [n][k]

  const int tid  = threadIdx.x;
  const int lane = tid & 31;
  const int wid  = tid >> 5;
  const int wm   = wid & 3;          // wave row group (0..3) -> 32 rows each
  const int wn   = wid >> 2;         // wave col group (0..1) -> 32 cols each
  const int lh   = lane & 15;
  const bool hiHalf = lane >= 16;

  const int m0 = blockIdx.x * 128;
  const int n1 = blockIdx.y * 64;

  f32x8 acc[2][2] = {};

  // staging assignments
  const int ar  = tid >> 1;          // A tile row 0..127
  const int ac  = (tid & 1) * 16;    // A tile col 0 or 16
  const int bn  = tid >> 2;          // B tile n 0..63
  const int bc0 = (tid & 3) * 8;     // B tile k offset 0,8,16,24 (16B chunks)

  // per-thread LDS destination (byte address) for the async B copy
  const unsigned sB_lds = (unsigned)(uintptr_t)(&sB[bn * 32 + bc0]);

  for (int kt = 0; kt < K; kt += 32) {
    // ---- stage B tile via async global->LDS (16 bytes per lane) ----
    {
      const bf16_t* gp = &WT[(size_t)(n1 + bn) * K + kt + bc0];
      asm volatile("global_load_async_to_lds_b128 %0, %1, off"
                   :: "v"(sB_lds), "v"(gp)
                   : "memory");
    }
    // ---- stage A tile (fp32 -> bf16) ----
    {
      float va[16];
      int grow = m0 + ar;
      if (grow < M) {
        const float4* p = (const float4*)&A[(size_t)grow * K + kt + ac];
#pragma unroll
        for (int q = 0; q < 4; ++q) {
          float4 f = p[q];
          va[q * 4 + 0] = f.x; va[q * 4 + 1] = f.y;
          va[q * 4 + 2] = f.z; va[q * 4 + 3] = f.w;
        }
      } else {
#pragma unroll
        for (int q = 0; q < 16; ++q) va[q] = 0.0f;
      }
#pragma unroll
      for (int q = 0; q < 16; ++q) sA[ar * 32 + ac + q] = (bf16_t)va[q];
    }
    // drain this wave's async copies, then workgroup barrier
    asm volatile("s_wait_asynccnt 0x0" ::: "memory");
    __syncthreads();

    // ---- load fragments ----
    bf16x16 af[2], bfr[2];
#pragma unroll
    for (int i = 0; i < 2; ++i) {
      int r  = wm * 32 + i * 16 + lh;
      int ka = hiHalf ? 8 : 0;                 // lanes 0-15: K 0-7/16-23; 16-31: K 8-15/24-31
      bf16x8 lo  = *(const bf16x8*)&sA[r * 32 + ka];
      bf16x8 hi8 = *(const bf16x8*)&sA[r * 32 + ka + 16];
      af[i] = __builtin_shufflevector(lo, hi8, 0, 1, 2, 3, 4, 5, 6, 7,
                                              8, 9, 10, 11, 12, 13, 14, 15);
    }
#pragma unroll
    for (int j = 0; j < 2; ++j) {
      int n  = wn * 32 + j * 16 + lh;
      int kb = hiHalf ? 8 : 0;                 // lanes 0-15: K 0-7/16-23; 16-31: K 8-15/24-31
      bf16x8 blo = *(const bf16x8*)&sB[n * 32 + kb];
      bf16x8 bhi = *(const bf16x8*)&sB[n * 32 + kb + 16];
      bfr[j] = __builtin_shufflevector(blo, bhi, 0, 1, 2, 3, 4, 5, 6, 7,
                                                 8, 9, 10, 11, 12, 13, 14, 15);
    }

    // ---- 2x2 WMMA ----
#pragma unroll
    for (int i = 0; i < 2; ++i)
#pragma unroll
      for (int j = 0; j < 2; ++j)
        acc[i][j] = __builtin_amdgcn_wmma_f32_16x16x32_bf16(
            false, af[i], false, bfr[j], (short)0, acc[i][j], false, false);

    __syncthreads();
  }

  // ---- epilogue: bias + optional ReLU; C/D layout: lane<16 -> M=v, lane>=16 -> M=v+8 ----
#pragma unroll
  for (int i = 0; i < 2; ++i) {
    int rbase = m0 + wm * 32 + i * 16 + (hiHalf ? 8 : 0);
#pragma unroll
    for (int j = 0; j < 2; ++j) {
      int col = n1 + wn * 32 + j * 16 + lh;
      float bv = bias[col];
#pragma unroll
      for (int v = 0; v < 8; ++v) {
        int row = rbase + v;
        if (row < M) {
          float val = acc[i][j][v] + bv;
          if (RELU) val = fmaxf(val, 0.0f);
          C[(size_t)row * N + col] = val;
        }
      }
    }
  }
}

// ---------------------------------------------------------------------------
// Host orchestration
// ---------------------------------------------------------------------------
extern "C" void kernel_launch(void* const* d_in, const int* in_sizes, int n_in,
                              void* d_out, int out_size, void* d_ws, size_t ws_size,
                              hipStream_t stream) {
  (void)n_in; (void)ws_size;
  const int Nn = in_sizes[0] / 11;     // 100000 nodes
  const int E  = in_sizes[2] / 2;      // 200000 edges
  const int H  = 512;
  const int G  = out_size;             // 4096 graphs

  const float* x   = (const float*)d_in[0];
  const float* pos = (const float*)d_in[1];
  const int*   ei  = (const int*)d_in[2];
  const int*   bat = (const int*)d_in[3];
  const int*   src = ei;
  const int*   dst = ei + E;
  const float *W1a = (const float*)d_in[5],  *b1a = (const float*)d_in[6];
  const float *W1b = (const float*)d_in[7],  *b1b = (const float*)d_in[8];
  const float *g1  = (const float*)d_in[9],  *be1 = (const float*)d_in[10];
  const float *W2a = (const float*)d_in[11], *b2a = (const float*)d_in[12];
  const float *W2b = (const float*)d_in[13], *b2b = (const float*)d_in[14];
  const float *g2  = (const float*)d_in[15], *be2 = (const float*)d_in[16];
  const float *W3a = (const float*)d_in[17], *b3a = (const float*)d_in[18];
  const float *W3b = (const float*)d_in[19], *b3b = (const float*)d_in[20];
  const float *g3  = (const float*)d_in[21], *be3 = (const float*)d_in[22];
  const float *Wf1 = (const float*)d_in[23], *bf1 = (const float*)d_in[24];
  const float *Wf2 = (const float*)d_in[25], *bf2 = (const float*)d_in[26];
  float* out = (float*)d_out;

  // workspace carving
  char* ws = (char*)d_ws;
  size_t off = 0;
  auto alloc = [&](size_t bytes) -> void* {
    void* p = ws + off;
    off += (bytes + 255) & ~(size_t)255;
    return p;
  };
  float*  bufH  = (float*)alloc((size_t)Nn * H * 4);
  float*  bufZ  = (float*)alloc((size_t)Nn * H * 4);
  float*  bufT  = (float*)alloc((size_t)Nn * H * 4);
  float*  h0    = (float*)alloc((size_t)Nn * 32 * 4);
  float*  z0    = (float*)alloc((size_t)Nn * 32 * 4);
  bf16_t* w1aT  = (bf16_t*)alloc((size_t)H * 32 * 2);   // [512 x 32]
  bf16_t* w1bT  = (bf16_t*)alloc((size_t)H * H * 2);    // [N x K] transposed
  bf16_t* w2aT  = (bf16_t*)alloc((size_t)H * H * 2);
  bf16_t* w2bT  = (bf16_t*)alloc((size_t)H * H * 2);
  bf16_t* w3aT  = (bf16_t*)alloc((size_t)H * H * 2);
  bf16_t* w3bT  = (bf16_t*)alloc((size_t)H * H * 2);
  bf16_t* wf1T  = (bf16_t*)alloc((size_t)H * H * 2);
  float*  scale = (float*)alloc(H * 4);
  float*  shift = (float*)alloc(H * 4);
  float*  sums  = (float*)alloc((size_t)G * H * 4);
  float*  cnts  = (float*)alloc((size_t)G * 4);
  float*  pooled= (float*)alloc((size_t)G * H * 4);
  float*  qbuf  = (float*)alloc((size_t)G * H * 4);

  const int TB = 256;
  auto blocks = [](long long n, int tb) { return (unsigned)((n + tb - 1) / tb); };

  // --- weight conversion: fp32 [K x N] -> bf16 transposed [N x K] ---
  cvt_pad_w1a_T<<<blocks(512 * 32, TB), TB, 0, stream>>>(W1a, w1aT);
  cvt_f32_bf16_T<<<blocks(H * H, TB), TB, 0, stream>>>(W1b, w1bT, H, H);
  cvt_f32_bf16_T<<<blocks(H * H, TB), TB, 0, stream>>>(W2a, w2aT, H, H);
  cvt_f32_bf16_T<<<blocks(H * H, TB), TB, 0, stream>>>(W2b, w2bT, H, H);
  cvt_f32_bf16_T<<<blocks(H * H, TB), TB, 0, stream>>>(W3a, w3aT, H, H);
  cvt_f32_bf16_T<<<blocks(H * H, TB), TB, 0, stream>>>(W3b, w3bT, H, H);
  cvt_f32_bf16_T<<<blocks(H * H, TB), TB, 0, stream>>>(Wf1, wf1T, H, H);

  dim3 gemmGrid((Nn + 127) / 128, H / 64);
  dim3 headGrid((G + 127) / 128, H / 64);

  // --- layer 1 (K = 32, padded) ---
  concat_pad<<<blocks((long long)Nn * 32, TB), TB, 0, stream>>>(x, pos, h0, Nn);
  copy_f32_vec4<<<blocks((long long)Nn * 8, TB), TB, 0, stream>>>(h0, z0, Nn * 8);
  edge_scatter<<<blocks((long long)E * 8, TB), TB, 0, stream>>>(h0, src, dst, z0, E, 32);
  gemm_bf16_wmma<true><<<gemmGrid, TB, 0, stream>>>(z0, w1aT, b1a, bufT, Nn, 32, H);
  gemm_bf16_wmma<false><<<gemmGrid, TB, 0, stream>>>(bufT, w1bT, b1b, bufZ, Nn, H, H);
  bn_stats<<<H, TB, 0, stream>>>(bufZ, g1, be1, scale, shift, Nn, H);
  bn_apply_relu<<<blocks((long long)Nn * H, TB), TB, 0, stream>>>(bufZ, scale, shift, bufH, Nn * H);

  // --- layer 2 ---
  edge_scatter<<<blocks((long long)E * 128, TB), TB, 0, stream>>>(bufH, src, dst, bufZ, E, H);
  gemm_bf16_wmma<true><<<gemmGrid, TB, 0, stream>>>(bufZ, w2aT, b2a, bufT, Nn, H, H);
  gemm_bf16_wmma<false><<<gemmGrid, TB, 0, stream>>>(bufT, w2bT, b2b, bufZ, Nn, H, H);
  bn_stats<<<H, TB, 0, stream>>>(bufZ, g2, be2, scale, shift, Nn, H);
  bn_apply_relu<<<blocks((long long)Nn * H, TB), TB, 0, stream>>>(bufZ, scale, shift, bufH, Nn * H);

  // --- layer 3 ---
  edge_scatter<<<blocks((long long)E * 128, TB), TB, 0, stream>>>(bufH, src, dst, bufZ, E, H);
  gemm_bf16_wmma<true><<<gemmGrid, TB, 0, stream>>>(bufZ, w3aT, b3a, bufT, Nn, H, H);
  gemm_bf16_wmma<false><<<gemmGrid, TB, 0, stream>>>(bufT, w3bT, b3b, bufZ, Nn, H, H);
  bn_stats<<<H, TB, 0, stream>>>(bufZ, g3, be3, scale, shift, Nn, H);
  bn_apply_relu<<<blocks((long long)Nn * H, TB), TB, 0, stream>>>(bufZ, scale, shift, bufH, Nn * H);

  // --- global mean pool ---
  zero_f32<<<blocks((long long)G * H, TB), TB, 0, stream>>>(sums, G * H);
  zero_f32<<<blocks(G, TB), TB, 0, stream>>>(cnts, G);
  pool_count<<<blocks(Nn, TB), TB, 0, stream>>>(bat, cnts, Nn);
  pool_sum<<<blocks((long long)Nn * 128, TB), TB, 0, stream>>>(bufH, bat, sums, Nn);
  pool_div<<<blocks((long long)G * H, TB), TB, 0, stream>>>(sums, cnts, pooled, G * H);

  // --- head ---
  gemm_bf16_wmma<true><<<headGrid, TB, 0, stream>>>(pooled, wf1T, bf1, qbuf, G, H, H);
  head2<<<G, 128, 0, stream>>>(qbuf, Wf2, bf2, out);
}
